// EGATClassifier_10471130267740
// MI455X (gfx1250) — matrix operand order repaired and checked by reading, not compiled, not run on hardware
//
#include <hip/hip_runtime.h>
#include <hip/hip_bf16.h>

#define N_NODES 50000
#define N_EDGES 800000
#define N_GRAPHS 500
#define IN_F 128
#define OUT_F 64
#define HEADS 4
#define HD 256            // HEADS*OUT_F
#define NEG_SLOPE 0.2f
#define EPS_GN 1e-6f
#define SW_STRIDE 72      // 64 k-values + 8 pad (16B-aligned rows, bank-spread)
#define SA_STRIDE 132     // 128 k-values + 4 pad floats (16B-aligned, bank-spread)

#if defined(__has_builtin)
#if __has_builtin(__builtin_amdgcn_global_load_async_to_lds_b128) && \
    __has_builtin(__builtin_amdgcn_s_wait_asynccnt)
#define HAS_ASYNC_LDS 1
#endif
#endif

typedef __attribute__((ext_vector_type(16))) __bf16 v16bf;
typedef __attribute__((ext_vector_type(8)))  __bf16 v8bf;
typedef __attribute__((ext_vector_type(8)))  float  v8f;
typedef __attribute__((ext_vector_type(4)))  int    v4i;
typedef __attribute__((address_space(1))) v4i* gv4i_p;   // global-AS vector pointer
typedef __attribute__((address_space(3))) v4i* lv4i_p;   // LDS-AS vector pointer

static __device__ __forceinline__ __bf16 f2bf(float f) {
  unsigned u = __float_as_uint(f);
  u += 0x7fffu + ((u >> 16) & 1u);              // round-to-nearest-even
  unsigned short s = (unsigned short)(u >> 16);
  return __builtin_bit_cast(__bf16, s);
}

static __device__ __forceinline__ unsigned encf(float f) {
  unsigned u = __float_as_uint(f);
  return (u & 0x80000000u) ? ~u : (u | 0x80000000u);
}
static __device__ __forceinline__ float decf(unsigned u) {
  u = (u & 0x80000000u) ? (u & 0x7fffffffu) : ~u;
  return __uint_as_float(u);
}

static __device__ __forceinline__ void pack4(v16bf& dst, int base, float4 f) {
  dst[base + 0] = f2bf(f.x); dst[base + 1] = f2bf(f.y);
  dst[base + 2] = f2bf(f.z); dst[base + 3] = f2bf(f.w);
}

// ---------------------------------------------------------------- init
__global__ __launch_bounds__(256) void k_init(float* rst, unsigned* emax, float* denom,
                                              float* gsum, float* vsum, float* pool, float* cnt) {
  long long i = (long long)blockIdx.x * 256 + threadIdx.x;
  if (i < (long long)N_NODES * HD) rst[i] = 0.0f;
  if (i < N_NODES * HEADS) { emax[i] = 0x00800000u /* enc(-FLT_MAX) */; denom[i] = 0.0f; }
  if (i < N_GRAPHS * OUT_F) { gsum[i] = 0.0f; vsum[i] = 0.0f; pool[i] = 0.0f; }
  if (i < N_GRAPHS) cnt[i] = 0.0f;
}

// ---------------------------------------------------------------- WMMA GEMM: feat = nfeats @ W
// Block: 256 threads = 8 waves, covers 32 rows x 256 cols.
// Wave w: row-tile rt = w&1, col range nc = (w>>1)*64, four 16x16 tiles -> 16 WMMAs/wave.
// A tile (32x128 f32) staged to LDS via GLOBAL_LOAD_ASYNC_TO_LDS_B128 (ASYNCcnt path).
// W staged per 64-k chunk, converted to bf16, transposed [n][k] in LDS.
__global__ __launch_bounds__(256) void k_gemm(const float* __restrict__ A,
                                              const float* __restrict__ W,
                                              float* __restrict__ feat) {
  __shared__ __bf16 sW[HD * SW_STRIDE];           // 36,864 B
  __shared__ float  sA[32 * SA_STRIDE];           // 16,896 B
  const int tid  = threadIdx.x;
  const int lane = tid & 31;
  const int wave = tid >> 5;
  const int ln   = lane & 15;
  const int hi   = lane >> 4;                     // lane group 0/1
  const int m0   = blockIdx.x * 32;
  const int rt   = wave & 1;                      // row tile within block
  const int nc   = (wave >> 1) * 64;              // col base (4 tiles of 16)
  const int kbA  = hi * 8;                        // A-frag k sub-base (ISA 16-bit A layout)
  const int kbB  = hi * 8;                        // B-frag k sub-base

  // ---- stage A tile: rows m0..m0+31, all 128 k, 16B pieces (32 per row) ----
  for (int i = tid; i < 1024; i += 256) {
    int pr = i >> 5;                              // row within tile
    int pc = i & 31;                              // 16B piece within row
    int gr = m0 + pr; if (gr >= N_NODES) gr = N_NODES - 1;
    const float* gp = A + (size_t)gr * IN_F + pc * 4;
    float*       lp = &sA[pr * SA_STRIDE + pc * 4];
#if HAS_ASYNC_LDS
    __builtin_amdgcn_global_load_async_to_lds_b128((gv4i_p)gp, (lv4i_p)lp, 0, 0);
#else
    *(float4*)lp = *(const float4*)gp;
#endif
  }
#if HAS_ASYNC_LDS
  __builtin_amdgcn_s_wait_asynccnt(0);
#endif

  v8f c0 = {}; v8f c1 = {}; v8f c2 = {}; v8f c3 = {};

  for (int chunk = 0; chunk < 2; ++chunk) {
    const int kc = chunk * 64;
    __syncthreads();                              // covers A-async completion + sW reuse
    // stage W[kc..kc+63][0..255] -> sW[n][k] (bf16, transposed)
    for (int i = tid; i < 64 * HD; i += 256) {
      int k = i >> 8;                             // 0..63
      int n = i & 255;
      sW[n * SW_STRIDE + k] = f2bf(W[(kc + k) * HD + n]);
    }
    __syncthreads();

    #pragma unroll
    for (int k0 = 0; k0 < 64; k0 += 32) {
      // A fragment from LDS: runs [k0+kbA..+7] and [k0+16+kbA..+7]
      const float* apl = &sA[(rt * 16 + ln) * SA_STRIDE + kc + k0 + kbA];
      float4 f0 = *(const float4*)(apl);
      float4 f1 = *(const float4*)(apl + 4);
      float4 g0 = *(const float4*)(apl + 16);
      float4 g1 = *(const float4*)(apl + 20);
      v16bf a;
      pack4(a, 0, f0); pack4(a, 4, f1); pack4(a, 8, g0); pack4(a, 12, g1);

      // 4 col tiles: B frags are runs [k0+kbB..+7] and [k0+16+kbB..+7] per lane (two aligned 16B ds loads)
      #pragma unroll
      for (int j = 0; j < 4; ++j) {
        const __bf16* bp = &sW[(nc + j * 16 + ln) * SW_STRIDE + k0 + kbB];
        v8bf blo = *(const v8bf*)(bp);
        v8bf bhi = *(const v8bf*)(bp + 16);
        v16bf b = __builtin_shufflevector(blo, bhi, 0,1,2,3,4,5,6,7,8,9,10,11,12,13,14,15);
        if (j == 0) c0 = __builtin_amdgcn_wmma_f32_16x16x32_bf16(false, a, false, b, (short)0, c0, false, false);
        if (j == 1) c1 = __builtin_amdgcn_wmma_f32_16x16x32_bf16(false, a, false, b, (short)0, c1, false, false);
        if (j == 2) c2 = __builtin_amdgcn_wmma_f32_16x16x32_bf16(false, a, false, b, (short)0, c2, false, false);
        if (j == 3) c3 = __builtin_amdgcn_wmma_f32_16x16x32_bf16(false, a, false, b, (short)0, c3, false, false);
      }
    }
  }

  // store D: VGPR v -> row m0 + rt*16 + v + 8*hi
  #pragma unroll
  for (int v = 0; v < 8; ++v) {
    int m = m0 + rt * 16 + v + 8 * hi;
    if (m < N_NODES) {
      float* fp = feat + (long long)m * HD + nc + ln;
      fp[0]  = c0[v];
      fp[16] = c1[v];
      fp[32] = c2[v];
      fp[48] = c3[v];
    }
  }
}

// ---------------------------------------------------------------- el/er node scores
__global__ __launch_bounds__(256) void k_scores(const float* __restrict__ feat,
                                                const float* __restrict__ attn_l,
                                                const float* __restrict__ attn_r,
                                                float* el, float* er) {
  int i = blockIdx.x * 256 + threadIdx.x;         // over N*HEADS
  if (i >= N_NODES * HEADS) return;
  int node = i >> 2, h = i & 3;
  const float* fp = feat + (long long)node * HD + h * OUT_F;
  const float* al = attn_l + h * OUT_F;
  const float* ar = attn_r + h * OUT_F;
  float sl = 0.0f, sr = 0.0f;
  #pragma unroll
  for (int d = 0; d < OUT_F; d += 4) {
    float4 f = *(const float4*)(fp + d);
    float4 l = *(const float4*)(al + d);
    float4 r = *(const float4*)(ar + d);
    sl += f.x * l.x + f.y * l.y + f.z * l.z + f.w * l.w;
    sr += f.x * r.x + f.y * r.y + f.z * r.z + f.w * r.w;
  }
  el[i] = sl; er[i] = sr;
}

// ---------------------------------------------------------------- edge pass 1: leaky-relu + segment max
__global__ __launch_bounds__(256) void k_edge1(const float* __restrict__ el, const float* __restrict__ er,
                                               const int* __restrict__ src, const int* __restrict__ dst,
                                               float* eedge, unsigned* emax) {
  int e = blockIdx.x * 256 + threadIdx.x;
  if (e >= N_EDGES) return;
  int s = src[e], d = dst[e];
  #pragma unroll
  for (int h = 0; h < HEADS; ++h) {
    float v = el[s * HEADS + h] + er[d * HEADS + h];
    v = v > 0.0f ? v : NEG_SLOPE * v;
    eedge[(long long)e * HEADS + h] = v;
    atomicMax(&emax[d * HEADS + h], encf(v));
  }
}

// ---------------------------------------------------------------- edge pass 2: exp + segment sum
__global__ __launch_bounds__(256) void k_edge2(const int* __restrict__ dst, const unsigned* __restrict__ emax,
                                               float* eedge, float* denom) {
  int e = blockIdx.x * 256 + threadIdx.x;
  if (e >= N_EDGES) return;
  int d = dst[e];
  #pragma unroll
  for (int h = 0; h < HEADS; ++h) {
    float m  = decf(emax[d * HEADS + h]);
    float ex = __expf(eedge[(long long)e * HEADS + h] - m);
    eedge[(long long)e * HEADS + h] = ex;
    unsafeAtomicAdd(&denom[d * HEADS + h], ex);
  }
}

// ---------------------------------------------------------------- edge pass 3: weighted message scatter
// 64 threads per edge; thread t handles feature d=t across 4 heads.
__global__ __launch_bounds__(256) void k_scatter(const float* __restrict__ feat,
                                                 const float* __restrict__ eedge,
                                                 const float* __restrict__ denom,
                                                 const int* __restrict__ src, const int* __restrict__ dst,
                                                 float* rst, float* attn_out) {
  int e = blockIdx.x * 4 + (threadIdx.x >> 6);
  int t = threadIdx.x & 63;
  if (e >= N_EDGES) return;
  int s = src[e], d = dst[e];
  float a[HEADS];
  #pragma unroll
  for (int h = 0; h < HEADS; ++h)
    a[h] = eedge[(long long)e * HEADS + h] / fmaxf(denom[d * HEADS + h], 1e-9f);
  if (t == 0)
    attn_out[e] = 0.25f * (a[0] + a[1] + a[2] + a[3]);
  #pragma unroll
  for (int h = 0; h < HEADS; ++h)
    unsafeAtomicAdd(&rst[(long long)d * HD + h * OUT_F + t],
                    feat[(long long)s * HD + h * OUT_F + t] * a[h]);
}

// ---------------------------------------------------------------- head mean + per-graph sum
__global__ __launch_bounds__(256) void k_hmean(const float* __restrict__ rst,
                                               const float* __restrict__ gat_bias,
                                               const int* __restrict__ node_batch,
                                               float* hbuf, float* gsum, float* cnt) {
  int i = blockIdx.x * 256 + threadIdx.x;         // over N*OUT_F
  if (i >= N_NODES * OUT_F) return;
  int n = i >> 6, d = i & 63;
  float acc = 0.0f;
  #pragma unroll
  for (int h = 0; h < HEADS; ++h)
    acc += rst[(long long)n * HD + h * OUT_F + d] + gat_bias[h * OUT_F + d];
  acc *= 0.25f;
  hbuf[i] = acc;
  int g = node_batch[n];
  unsafeAtomicAdd(&gsum[g * OUT_F + d], acc);
  if (d == 0) unsafeAtomicAdd(&cnt[g], 1.0f);
}

// ---------------------------------------------------------------- GraphNorm pass 2: center + variance
__global__ __launch_bounds__(256) void k_subvar(const int* __restrict__ node_batch,
                                                const float* __restrict__ gsum, const float* __restrict__ cnt,
                                                const float* __restrict__ gn_mean_scale,
                                                float* hbuf, float* vsum) {
  int i = blockIdx.x * 256 + threadIdx.x;
  if (i >= N_NODES * OUT_F) return;
  int n = i >> 6, d = i & 63;
  int g = node_batch[n];
  float cm   = fmaxf(cnt[g], 1.0f);
  float mean = gsum[g * OUT_F + d] / cm;
  float sub  = hbuf[i] - mean * gn_mean_scale[d];
  hbuf[i] = sub;
  unsafeAtomicAdd(&vsum[g * OUT_F + d], sub * sub);
}

// ---------------------------------------------------------------- normalize + ELU + pool
__global__ __launch_bounds__(256) void k_norm(const int* __restrict__ node_batch,
                                              const float* __restrict__ vsum, const float* __restrict__ cnt,
                                              const float* __restrict__ gn_weight, const float* __restrict__ gn_bias,
                                              const float* __restrict__ hbuf,
                                              float* h_out, float* pool) {
  int i = blockIdx.x * 256 + threadIdx.x;
  if (i >= N_NODES * OUT_F) return;
  int n = i >> 6, d = i & 63;
  int g = node_batch[n];
  float cm  = fmaxf(cnt[g], 1.0f);
  float var = vsum[g * OUT_F + d] / cm;
  float hn  = gn_weight[d] * hbuf[i] * rsqrtf(var + EPS_GN) + gn_bias[d];
  float ho  = hn > 0.0f ? hn : (__expf(hn) - 1.0f);
  h_out[i] = ho;
  unsafeAtomicAdd(&pool[g * OUT_F + d], ho);
}

// ---------------------------------------------------------------- classify
__global__ __launch_bounds__(256) void k_cls(const float* __restrict__ pool, const float* __restrict__ cnt,
                                             const float* __restrict__ cls_W, const float* __restrict__ cls_b,
                                             float* logits) {
  int g = blockIdx.x * 256 + threadIdx.x;
  if (g >= N_GRAPHS) return;
  float cm = fmaxf(cnt[g], 1.0f);
  float acc = 0.0f;
  #pragma unroll 4
  for (int d = 0; d < OUT_F; ++d)
    acc += pool[g * OUT_F + d] * cls_W[d];
  logits[g] = acc / cm + cls_b[0];
}

// ================================================================ host
extern "C" void kernel_launch(void* const* d_in, const int* in_sizes, int n_in,
                              void* d_out, int out_size, void* d_ws, size_t ws_size,
                              hipStream_t stream) {
  const float* nfeats       = (const float*)d_in[0];
  const float* W            = (const float*)d_in[1];
  const float* attn_l       = (const float*)d_in[2];
  const float* attn_r       = (const float*)d_in[3];
  const float* gat_bias     = (const float*)d_in[4];
  const float* gn_weight    = (const float*)d_in[5];
  const float* gn_bias      = (const float*)d_in[6];
  const float* gn_mean_scale= (const float*)d_in[7];
  const float* cls_W        = (const float*)d_in[8];
  const float* cls_b        = (const float*)d_in[9];
  const int*   src          = (const int*)d_in[10];
  const int*   dst          = (const int*)d_in[11];
  const int*   node_batch   = (const int*)d_in[12];

  float* out    = (float*)d_out;
  float* logits = out;                      // G
  float* attn   = out + N_GRAPHS;           // E
  float* h_out  = out + N_GRAPHS + N_EDGES; // N*OUT_F

  float* ws = (float*)d_ws;
  size_t off = 0;
  float*    feat  = ws + off; off += (size_t)N_NODES * HD;
  float*    rst   = ws + off; off += (size_t)N_NODES * HD;
  float*    el    = ws + off; off += (size_t)N_NODES * HEADS;
  float*    er    = ws + off; off += (size_t)N_NODES * HEADS;
  float*    eedge = ws + off; off += (size_t)N_EDGES * HEADS;
  unsigned* emax  = (unsigned*)(ws + off); off += (size_t)N_NODES * HEADS;
  float*    denom = ws + off; off += (size_t)N_NODES * HEADS;
  float*    hbuf  = ws + off; off += (size_t)N_NODES * OUT_F;
  float*    gsum  = ws + off; off += (size_t)N_GRAPHS * OUT_F;
  float*    vsum  = ws + off; off += (size_t)N_GRAPHS * OUT_F;
  float*    pool  = ws + off; off += (size_t)N_GRAPHS * OUT_F;
  float*    cnt   = ws + off; off += (size_t)N_GRAPHS;
  (void)ws_size; (void)in_sizes; (void)n_in; (void)out_size;

  dim3 blk(256);
  k_init   <<<dim3((N_NODES * HD + 255) / 256), blk, 0, stream>>>(rst, emax, denom, gsum, vsum, pool, cnt);
  k_gemm   <<<dim3((N_NODES + 31) / 32), blk, 0, stream>>>(nfeats, W, feat);
  k_scores <<<dim3((N_NODES * HEADS + 255) / 256), blk, 0, stream>>>(feat, attn_l, attn_r, el, er);
  k_edge1  <<<dim3((N_EDGES + 255) / 256), blk, 0, stream>>>(el, er, src, dst, eedge, emax);
  k_edge2  <<<dim3((N_EDGES + 255) / 256), blk, 0, stream>>>(dst, emax, eedge, denom);
  k_scatter<<<dim3((N_EDGES + 3) / 4), blk, 0, stream>>>(feat, eedge, denom, src, dst, rst, attn);
  k_hmean  <<<dim3((N_NODES * OUT_F + 255) / 256), blk, 0, stream>>>(rst, gat_bias, node_batch, hbuf, gsum, cnt);
  k_subvar <<<dim3((N_NODES * OUT_F + 255) / 256), blk, 0, stream>>>(node_batch, gsum, cnt, gn_mean_scale, hbuf, vsum);
  k_norm   <<<dim3((N_NODES * OUT_F + 255) / 256), blk, 0, stream>>>(node_batch, vsum, cnt, gn_weight, gn_bias, hbuf, h_out, pool);
  k_cls    <<<dim3((N_GRAPHS + 255) / 256), blk, 0, stream>>>(pool, cnt, cls_W, cls_b, logits);
}
